// DenoiseGAT_3470333575474
// MI455X (gfx1250) — compile-verified, matrix operand order checked by reading, not run on hardware
//
#include <hip/hip_runtime.h>
#include <hip/hip_bf16.h>
#include <math.h>
#include <stdint.h>
#include <stddef.h>

// ---------------------------------------------------------------------------
// Problem constants
// ---------------------------------------------------------------------------
#define NPOLY 2048      // B
#define VV    64        // vertices per polygon
#define NN    (NPOLY*VV)
#define HIDC  256       // hidden width (4 heads * 64 = 256, layer3: 1*256)
#define K0PAD 160       // NIN=134 padded to multiple of 32

typedef __bf16 bf16_t;
typedef __attribute__((ext_vector_type(16))) __bf16    v16bf;
typedef __attribute__((ext_vector_type(8)))  float     v8f;
typedef __attribute__((ext_vector_type(4)))  uint32_t  v4u;

union FragBF { v4u u[2]; v16bf v; };

// ---------------------------------------------------------------------------
// Fragment loaders (contiguous k order used by this code; B packed the same)
// A 16x32 bf16: lanes 0-15 -> M=lane, K=k0..k0+15 ; lanes 16-31 -> M=lane-16,
// K=k0+16..k0+31. Row-major source, so each lane reads 32 contiguous bytes.
// ---------------------------------------------------------------------------
__device__ __forceinline__ v16bf load_frag_a(const bf16_t* __restrict__ base,
                                             int kpad, int m0, int k0, int lane) {
  const int row = m0 + (lane & 15);
  const int kk  = k0 + ((lane >> 4) << 4);
  FragBF f;
  const v4u* p = reinterpret_cast<const v4u*>(base + (size_t)row * kpad + kk);
  f.u[0] = p[0];
  f.u[1] = p[1];
  return f.v;
}

// B 32x16 bf16: pre-swizzled by pack_w_kernel so each lane's 16 values are
// 32 contiguous bytes at ((kt*NT + nt)*32 + lane)*16.
__device__ __forceinline__ v16bf load_frag_b(const bf16_t* __restrict__ wp,
                                             int kt, int nt, int NT, int lane) {
  FragBF f;
  const v4u* p = reinterpret_cast<const v4u*>(wp + ((size_t)(kt * NT + nt) * 32 + lane) * 16);
  f.u[0] = p[0];
  f.u[1] = p[1];
  return f.v;
}

__device__ __forceinline__ float lrelu02(float x) { return x > 0.0f ? x : 0.2f * x; }
__device__ __forceinline__ float elu1(float x)    { return x > 0.0f ? x : (__expf(x) - 1.0f); }
__device__ __forceinline__ float silu(float x)    { return x / (1.0f + __expf(-x)); }

// ---------------------------------------------------------------------------
// 1) time embedding: sinusoidal(128) @ tW (128x128) + tb, SiLU -> temb_act f32
// ---------------------------------------------------------------------------
__global__ __launch_bounds__(128) void temb_kernel(const int* __restrict__ t,
                                                   const float* __restrict__ tW,
                                                   const float* __restrict__ tb,
                                                   float* __restrict__ temb_act) {
  __shared__ float s_raw[128];
  const int b   = blockIdx.x;
  const int tid = threadIdx.x;
  const float tv = (float)t[b];
  if (tid < 64) {
    const float fr = __expf(-logf(10000.0f) * (float)tid / 63.0f);
    const float te = tv * fr;
    s_raw[tid]      = __sinf(te);
    s_raw[tid + 64] = __cosf(te);
  }
  __syncthreads();
  float s = tb[tid];
  #pragma unroll 4
  for (int k = 0; k < 128; k++) s += s_raw[k] * tW[k * 128 + tid];
  temb_act[b * 128 + tid] = silu(s);
}

// ---------------------------------------------------------------------------
// 2) build h0: [coords(2) | cycle pos-enc(4) | temb(128) | zero-pad(26)] bf16
// ---------------------------------------------------------------------------
__global__ void build_h0_kernel(const float* __restrict__ x,
                                const float* __restrict__ temb_act,
                                bf16_t* __restrict__ h0) {
  const size_t e = (size_t)blockIdx.x * blockDim.x + threadIdx.x;
  if (e >= (size_t)NN * K0PAD) return;
  const int    col = (int)(e % K0PAD);
  const size_t n   = e / K0PAD;
  const int b = (int)(n >> 6);
  const int v = (int)(n & 63);
  float val;
  if (col < 2) {
    val = x[n * 2 + col];
  } else if (col < 6) {
    const float ph = (float)v * (6.28318530717958647692f / 64.0f);
    const float m  = (col < 4) ? ph : 2.0f * ph;
    val = ((col & 1) == 0) ? __sinf(m) : __cosf(m);
  } else if (col < 134) {
    val = temb_act[b * 128 + (col - 6)];
  } else {
    val = 0.0f;
  }
  h0[e] = (bf16_t)val;
}

// ---------------------------------------------------------------------------
// 3) weight pack: f32 (Krows x Ncols) -> bf16 WMMA-B fragment order, K zero-pad
// ---------------------------------------------------------------------------
__global__ void pack_w_kernel(const float* __restrict__ W, bf16_t* __restrict__ dst,
                              int Krows, int Kpad, int Ncols) {
  const int e = blockIdx.x * blockDim.x + threadIdx.x;
  const int NT = Ncols >> 4;
  const int total = (Kpad >> 5) * NT * 512;
  if (e >= total) return;
  const int i    = e & 15;
  const int lane = (e >> 4) & 31;
  const int tile = e >> 9;
  const int nt   = tile % NT;
  const int kt   = tile / NT;
  const int row  = kt * 32 + ((lane >> 4) << 4) + i;
  const int col  = nt * 16 + (lane & 15);
  const float v  = (row < Krows) ? W[row * Ncols + col] : 0.0f;
  dst[e] = (bf16_t)v;
}

// ---------------------------------------------------------------------------
// 4) fused GAT layer: one workgroup per polygon (64 nodes, cycle-local).
//    proj = h@W via v_wmma_f32_16x16x32_bf16 -> LDS; per-(node,head) attention
//    scores + 3-way softmax; skip GEMM (WMMA) fused into combine+bias+ELU.
// ---------------------------------------------------------------------------
__global__ __launch_bounds__(256) void gat_layer_kernel(
    const bf16_t* __restrict__ hin, bf16_t* __restrict__ hout,
    const bf16_t* __restrict__ Wp, const bf16_t* __restrict__ Sp, int has_skip,
    const float* __restrict__ a_src, const float* __restrict__ a_tgt,
    const float* __restrict__ bias, int kpad, int nh, int act) {
  __shared__ bf16_t s_proj[64 * 256];     // 32 KB
  __shared__ float  s_ss[64 * 4];
  __shared__ float  s_st[64 * 4];
  __shared__ float  s_alpha[64 * 4 * 3];

  const int tid  = threadIdx.x;
  const int lane = tid & 31;
  const int wave = tid >> 5;
  const int poly = blockIdx.x;
  const bf16_t* hrow = hin + (size_t)poly * 64 * kpad;
  const int ktiles = kpad >> 5;
  const int fout   = 256 / nh;
  const int rt  = wave >> 1;              // row tile 0..3 (16 rows each)
  const int ct0 = (wave & 1) * 8;         // first of 8 col tiles

  // ---- proj GEMM: each wave -> one row tile x 8 col tiles (A reuse) ----
  v8f acc[8];
  #pragma unroll
  for (int i = 0; i < 8; i++)
    #pragma unroll
    for (int r = 0; r < 8; r++) acc[i][r] = 0.0f;

  for (int kt = 0; kt < ktiles; kt++) {
    const v16bf a = load_frag_a(hrow, kpad, rt * 16, kt * 32, lane);
    #pragma unroll
    for (int i = 0; i < 8; i++) {
      const v16bf b = load_frag_b(Wp, kt, ct0 + i, 16, lane);
      acc[i] = __builtin_amdgcn_wmma_f32_16x16x32_bf16(
          false, a, false, b, (short)0, acc[i], false, false);
    }
  }
  // C layout: VGPR r -> M = r (+8 for lanes 16-31), N = lane&15
  #pragma unroll
  for (int i = 0; i < 8; i++) {
    const int col = (ct0 + i) * 16 + (lane & 15);
    #pragma unroll
    for (int r = 0; r < 8; r++) {
      const int row = rt * 16 + r + ((lane >> 4) << 3);
      s_proj[row * 256 + col] = (bf16_t)acc[i][r];
    }
  }
  __syncthreads();

  // ---- attention scores s_src/s_tgt per (node, head) ----
  if (tid < 64 * nh) {
    const int j = tid & 63;
    const int h = tid >> 6;
    float a1 = 0.0f, a2 = 0.0f;
    for (int f = 0; f < fout; f++) {
      const float p = (float)s_proj[j * 256 + h * fout + f];
      a1 += p * a_src[h * fout + f];
      a2 += p * a_tgt[h * fout + f];
    }
    s_ss[j * 4 + h] = a1;
    s_st[j * 4 + h] = a2;
  }
  __syncthreads();

  // ---- 3-way softmax over {prev, next, self} incoming edges ----
  if (tid < 64 * nh) {
    const int j  = tid & 63;
    const int h  = tid >> 6;
    const int jp = (j + 63) & 63;
    const int jn = (j + 1) & 63;
    const float st = s_st[j * 4 + h];
    const float ep = lrelu02(s_ss[jp * 4 + h] + st);
    const float en = lrelu02(s_ss[jn * 4 + h] + st);
    const float es = lrelu02(s_ss[j * 4 + h] + st);
    const float m  = fmaxf(es, fmaxf(ep, en));
    const float xp = __expf(ep - m), xn = __expf(en - m), xs = __expf(es - m);
    const float d  = xp + xn + xs + 1e-16f;
    s_alpha[(j * 4 + h) * 3 + 0] = xp / d;
    s_alpha[(j * 4 + h) * 3 + 1] = xn / d;
    s_alpha[(j * 4 + h) * 3 + 2] = xs / d;
  }

  // ---- skip GEMM (h @ skip) into registers (WMMA again) ----
  v8f acc2[8];
  #pragma unroll
  for (int i = 0; i < 8; i++)
    #pragma unroll
    for (int r = 0; r < 8; r++) acc2[i][r] = 0.0f;
  if (has_skip) {
    for (int kt = 0; kt < ktiles; kt++) {
      const v16bf a = load_frag_a(hrow, kpad, rt * 16, kt * 32, lane);
      #pragma unroll
      for (int i = 0; i < 8; i++) {
        const v16bf b = load_frag_b(Sp, kt, ct0 + i, 16, lane);
        acc2[i] = __builtin_amdgcn_wmma_f32_16x16x32_bf16(
            false, a, false, b, (short)0, acc2[i], false, false);
      }
    }
  }
  __syncthreads();

  // ---- combine: out = skip + sum(alpha * proj[src]) + bias, optional ELU ----
  #pragma unroll
  for (int i = 0; i < 8; i++) {
    const int col = (ct0 + i) * 16 + (lane & 15);
    const int h   = col / fout;
    #pragma unroll
    for (int r = 0; r < 8; r++) {
      const int j  = rt * 16 + r + ((lane >> 4) << 3);
      const int jp = (j + 63) & 63;
      const int jn = (j + 1) & 63;
      const float skipv = has_skip ? acc2[i][r]
                                   : (float)hrow[(size_t)j * kpad + col];
      const float ap = s_alpha[(j * 4 + h) * 3 + 0];
      const float an = s_alpha[(j * 4 + h) * 3 + 1];
      const float as = s_alpha[(j * 4 + h) * 3 + 2];
      float v = skipv
              + ap * (float)s_proj[jp * 256 + col]
              + an * (float)s_proj[jn * 256 + col]
              + as * (float)s_proj[j  * 256 + col]
              + bias[col];
      if (act) v = elu1(v);
      hout[((size_t)poly * 64 + j) * 256 + col] = (bf16_t)v;
    }
  }
}

// ---------------------------------------------------------------------------
// 5) dense 64-row GEMM (h @ h1W) + bias + SiLU -> bf16 (pure WMMA, no LDS)
// ---------------------------------------------------------------------------
__global__ __launch_bounds__(256) void gemm64_bias_silu_kernel(
    const bf16_t* __restrict__ hin, bf16_t* __restrict__ hout,
    const bf16_t* __restrict__ Wp, const float* __restrict__ bias) {
  const int tid  = threadIdx.x;
  const int lane = tid & 31;
  const int wave = tid >> 5;
  const int poly = blockIdx.x;
  const bf16_t* hrow = hin + (size_t)poly * 64 * 256;
  const int rt  = wave >> 1;
  const int ct0 = (wave & 1) * 8;

  v8f acc[8];
  #pragma unroll
  for (int i = 0; i < 8; i++)
    #pragma unroll
    for (int r = 0; r < 8; r++) acc[i][r] = 0.0f;

  for (int kt = 0; kt < 8; kt++) {
    const v16bf a = load_frag_a(hrow, 256, rt * 16, kt * 32, lane);
    #pragma unroll
    for (int i = 0; i < 8; i++) {
      const v16bf b = load_frag_b(Wp, kt, ct0 + i, 16, lane);
      acc[i] = __builtin_amdgcn_wmma_f32_16x16x32_bf16(
          false, a, false, b, (short)0, acc[i], false, false);
    }
  }
  #pragma unroll
  for (int i = 0; i < 8; i++) {
    const int col = (ct0 + i) * 16 + (lane & 15);
    #pragma unroll
    for (int r = 0; r < 8; r++) {
      const int row = rt * 16 + r + ((lane >> 4) << 3);
      const float v = silu(acc[i][r] + bias[col]);
      hout[((size_t)poly * 64 + row) * 256 + col] = (bf16_t)v;
    }
  }
}

// ---------------------------------------------------------------------------
// 6) output head: (N x 256) bf16 @ (256 x 2) f32 + b -> f32 noise
// ---------------------------------------------------------------------------
__global__ void head_out_kernel(const bf16_t* __restrict__ h4,
                                const float* __restrict__ h2W,
                                const float* __restrict__ h2b,
                                float* __restrict__ out) {
  const size_t n = (size_t)blockIdx.x * blockDim.x + threadIdx.x;
  if (n >= (size_t)NN) return;
  const bf16_t* r = h4 + n * 256;
  float s0 = h2b[0], s1 = h2b[1];
  #pragma unroll 4
  for (int k = 0; k < 256; k++) {
    const float v = (float)r[k];
    s0 += v * h2W[k * 2 + 0];
    s1 += v * h2W[k * 2 + 1];
  }
  out[n * 2 + 0] = s0;
  out[n * 2 + 1] = s1;
}

// ---------------------------------------------------------------------------
// Host launcher
// ---------------------------------------------------------------------------
extern "C" void kernel_launch(void* const* d_in, const int* in_sizes, int n_in,
                              void* d_out, int out_size, void* d_ws, size_t ws_size,
                              hipStream_t stream) {
  (void)in_sizes; (void)n_in; (void)out_size; (void)ws_size;
  // setup_inputs() dict order:
  const float* x     = (const float*)d_in[0];
  const int*   t     = (const int*)  d_in[1];
  const float* tW    = (const float*)d_in[2];
  const float* tb    = (const float*)d_in[3];
  const float* W[4]    = {(const float*)d_in[4],  (const float*)d_in[8],
                          (const float*)d_in[12], (const float*)d_in[16]};
  const float* asrc[4] = {(const float*)d_in[5],  (const float*)d_in[9],
                          (const float*)d_in[13], (const float*)d_in[17]};
  const float* atgt[4] = {(const float*)d_in[6],  (const float*)d_in[10],
                          (const float*)d_in[14], (const float*)d_in[18]};
  const float* bi[4]   = {(const float*)d_in[7],  (const float*)d_in[11],
                          (const float*)d_in[15], (const float*)d_in[19]};
  const float* skip[3] = {(const float*)d_in[20], (const float*)d_in[21],
                          (const float*)d_in[22]};
  const float* h1W = (const float*)d_in[23];
  const float* h1b = (const float*)d_in[24];
  const float* h2W = (const float*)d_in[25];
  const float* h2b = (const float*)d_in[26];

  // workspace carve (256B aligned)
  char* ws = (char*)d_ws;
  size_t o = 0;
  auto carve = [&](size_t bytes) { void* p = ws + o; o += (bytes + 255) & ~(size_t)255; return p; };
  float*  temb_act = (float*) carve((size_t)NPOLY * 128 * 4);
  bf16_t* h0       = (bf16_t*)carve((size_t)NN * K0PAD * 2);
  bf16_t* hA       = (bf16_t*)carve((size_t)NN * 256 * 2);
  bf16_t* hB       = (bf16_t*)carve((size_t)NN * 256 * 2);
  bf16_t* w0p  = (bf16_t*)carve((size_t)5 * 16 * 512 * 2);
  bf16_t* s0p  = (bf16_t*)carve((size_t)5 * 16 * 512 * 2);
  bf16_t* w1p  = (bf16_t*)carve((size_t)8 * 16 * 512 * 2);
  bf16_t* s1p  = (bf16_t*)carve((size_t)8 * 16 * 512 * 2);
  bf16_t* w2p  = (bf16_t*)carve((size_t)8 * 16 * 512 * 2);
  bf16_t* s2p  = (bf16_t*)carve((size_t)8 * 16 * 512 * 2);
  bf16_t* w3p  = (bf16_t*)carve((size_t)8 * 16 * 512 * 2);
  bf16_t* h1wp = (bf16_t*)carve((size_t)8 * 16 * 512 * 2);

  // 1) time embedding
  temb_kernel<<<NPOLY, 128, 0, stream>>>(t, tW, tb, temb_act);
  // 2) node features
  {
    const size_t tot = (size_t)NN * K0PAD;
    build_h0_kernel<<<(unsigned)((tot + 255) / 256), 256, 0, stream>>>(x, temb_act, h0);
  }
  // 3) pack weights into WMMA-B fragment layout (bf16)
  auto pack = [&](const float* src, bf16_t* dst, int K, int Kpad, int Ncols) {
    const int total = (Kpad >> 5) * (Ncols >> 4) * 512;
    pack_w_kernel<<<(total + 255) / 256, 256, 0, stream>>>(src, dst, K, Kpad, Ncols);
  };
  pack(W[0],    w0p,  134, K0PAD, 256);
  pack(skip[0], s0p,  134, K0PAD, 256);
  pack(W[1],    w1p,  256, 256,   256);
  pack(skip[1], s1p,  256, 256,   256);
  pack(W[2],    w2p,  256, 256,   256);
  pack(skip[2], s2p,  256, 256,   256);
  pack(W[3],    w3p,  256, 256,   256);
  pack(h1W,     h1wp, 256, 256,   256);

  // 4) GAT layers (one workgroup per polygon; cycle neighbors are block-local)
  gat_layer_kernel<<<NPOLY, 256, 0, stream>>>(h0, hA, w0p, s0p, 1,
                                              asrc[0], atgt[0], bi[0], K0PAD, 4, 1);
  gat_layer_kernel<<<NPOLY, 256, 0, stream>>>(hA, hB, w1p, s1p, 1,
                                              asrc[1], atgt[1], bi[1], 256, 4, 1);
  gat_layer_kernel<<<NPOLY, 256, 0, stream>>>(hB, hA, w2p, s2p, 1,
                                              asrc[2], atgt[2], bi[2], 256, 4, 1);
  gat_layer_kernel<<<NPOLY, 256, 0, stream>>>(hA, hB, w3p, nullptr, 0,
                                              asrc[3], atgt[3], bi[3], 256, 1, 0);
  // 5) h1 dense + SiLU
  gemm64_bias_silu_kernel<<<NPOLY, 256, 0, stream>>>(hB, hA, h1wp, h1b);
  // 6) 256 -> 2 head, f32 output
  head_out_kernel<<<(NN + 255) / 256, 256, 0, stream>>>(hA, h2W, h2b, (float*)d_out);
}
